// BidirectionalAttentionContrastiveLoss_82824149336834
// MI455X (gfx1250) — compile-verified, matrix operand order checked by reading, not run on hardware
//
#include <hip/hip_runtime.h>
#include <hip/hip_bf16.h>
#include <math.h>

// ---------------------------------------------------------------------------
// BidirectionalAttentionContrastiveLoss for MI455X (gfx1250, wave32, WMMA).
// GEMM math -> v_wmma_f32_16x16x32_bf16; K staging -> global_load_async_to_lds.
// ---------------------------------------------------------------------------

typedef __attribute__((ext_vector_type(16))) __bf16 bf16x16;
typedef __attribute__((ext_vector_type(8)))  __bf16 bf16x8;
typedef __attribute__((ext_vector_type(4)))  __bf16 bf16x4;
typedef __attribute__((ext_vector_type(8)))  float  f32x8;

#define DIM   256
#define HEADS 4
#define HD    64
#define BATCH 32
#define NVIS  256
#define NLANG 64
#define SCALE 0.125f            // 1/sqrt(64)
#define TEMP  0.07f

// ---------------------------------------------------------------------------
// Fragment loader for 16-bit WMMA operands (A 16x32 layout per CDNA5 ISA:
// lane&15 = M row; lane>>4 selects which 8-wide K chunks the lane holds:
// elements 0..7 -> K = 8*half..8*half+7, elements 8..15 -> K = 16+8*half..+7).
// B (32x16) uses the mirrored layout with lane&15 = N column reading K^T rows.
// base must be 16B aligned; works for global or LDS pointers.
// ---------------------------------------------------------------------------
__device__ __forceinline__ bf16x16 load_frag(const __bf16* base, int ld,
                                             int row, int kbase, int kh) {
    const __bf16* p = base + (size_t)row * ld + kbase + kh * 8;
    bf16x8 lo = *(const bf16x8*)(p);
    bf16x8 hi = *(const bf16x8*)(p + 16);
    bf16x16 f;
#pragma unroll
    for (int e = 0; e < 8; ++e) { f[e] = lo[e]; f[e + 8] = hi[e]; }
    return f;
}

__device__ __forceinline__ f32x8 wmma_bf16(bf16x16 a, bf16x16 b, f32x8 c) {
    return __builtin_amdgcn_wmma_f32_16x16x32_bf16(
        false, a, false, b, (short)0, c, false, false);
}

// ------------------------- sorted top-3 helpers ----------------------------
struct Top3 { float a, b, c; };

__device__ __forceinline__ void t3_init(Top3& t) {
    t.a = -INFINITY; t.b = -INFINITY; t.c = -INFINITY;
}
__device__ __forceinline__ void t3_insert(Top3& t, float v) {
    if (v > t.a)      { t.c = t.b; t.b = t.a; t.a = v; }
    else if (v > t.b) { t.c = t.b; t.b = v; }
    else if (v > t.c) { t.c = v; }
}

// ---------------------------------------------------------------------------
// Kernel 0: convert Wq|Wk (first 2*D rows of in_proj_weight) to bf16.
// ---------------------------------------------------------------------------
__global__ void cvt_weights_kernel(const float* __restrict__ W,
                                   __bf16* __restrict__ Wbf, int n) {
    int i = blockIdx.x * blockDim.x + threadIdx.x;
    if (i < n) Wbf[i] = (__bf16)W[i];
}

// ---------------------------------------------------------------------------
// Kernel 1: projections. One block per 16-row token tile (vis tiles first,
// then lang tiles). Block = 128 threads (4 waves). X tile staged to LDS as
// bf16 with +8 element row pad (conflict-free ds_load_b128 fragment reads).
// Each wave covers d-tiles {w, w+4, w+8, w+12} for both Wq and Wk.
// ---------------------------------------------------------------------------
__global__ void proj_kernel(const float* __restrict__ lang,
                            const float* __restrict__ vis,
                            const float* __restrict__ bias,
                            const __bf16* __restrict__ Wbf,
                            __bf16* __restrict__ Qv, __bf16* __restrict__ Kv,
                            __bf16* __restrict__ Ql, __bf16* __restrict__ Kl) {
    const int bx = blockIdx.x;
    const float* X; __bf16 *Qo, *Ko; int N, batch, tile;
    if (bx < BATCH * (NVIS / 16)) {           // 512 vis tiles
        X = vis;  Qo = Qv; Ko = Kv; N = NVIS;  batch = bx >> 4; tile = bx & 15;
    } else {                                   // 128 lang tiles
        int t = bx - BATCH * (NVIS / 16);
        X = lang; Qo = Ql; Ko = Kl; N = NLANG; batch = t >> 2;  tile = t & 3;
    }

    __shared__ __bf16 lx[16 * 264];            // 16 x (256+8) bf16

    const int tid = threadIdx.x;
    // stage + f32->bf16 convert: 16 rows x 64 float4 chunks (VALU convert,
    // so this path cannot use the async-LDS copy)
    for (int idx = tid; idx < 16 * 64; idx += 128) {
        int row = idx >> 6, c4 = idx & 63;
        const float4 v = *(const float4*)(X +
            ((size_t)(batch * N + tile * 16 + row)) * DIM + c4 * 4);
        bf16x4 bv = { (__bf16)v.x, (__bf16)v.y, (__bf16)v.z, (__bf16)v.w };
        *(bf16x4*)&lx[row * 264 + c4 * 4] = bv;
    }
    __syncthreads();

    const int wave = tid >> 5, lane = tid & 31;
    const int l16 = lane & 15, kh = lane >> 4;

    // Hoist all 8 A fragments (K = 0..255 in steps of 32), reused 8x each.
    bf16x16 A[8];
#pragma unroll
    for (int ks = 0; ks < 8; ++ks)
        A[ks] = load_frag(lx, 264, l16, ks * 32, kh);

    for (int dt = wave; dt < 16; dt += 4) {
#pragma unroll
        for (int wq = 0; wq < 2; ++wq) {       // 0 = Wq/Q, 1 = Wk/K
            const __bf16* wb = Wbf + (size_t)wq * DIM * DIM;
            const float bv = bias[wq * DIM + dt * 16 + l16];
            f32x8 c;
#pragma unroll
            for (int r = 0; r < 8; ++r) c[r] = bv;   // bias on output col N
#pragma unroll
            for (int ks = 0; ks < 8; ++ks) {
                bf16x16 b = load_frag(wb, DIM, dt * 16 + l16, ks * 32, kh);
                c = wmma_bf16(A[ks], b, c);
            }
            __bf16* out = wq ? Ko : Qo;
#pragma unroll
            for (int r = 0; r < 8; ++r) {
                int M = r + 8 * kh;            // C-frag: vgpr r -> row r / r+8
                out[((size_t)(batch * N + tile * 16 + M)) * DIM
                    + dt * 16 + l16] = (__bf16)c[r];
            }
        }
    }
}

// ---------------------------------------------------------------------------
// Kernel 2: pairwise attention similarity. One block per batch pair (i,j).
// Per head: stage target-K head slice to LDS with GLOBAL_LOAD_ASYNC_TO_LDS
// (ASYNCcnt-tracked, no VGPR round trip), each wave computes S = Q K^T for
// its 16-row query tiles via WMMA, then softmax over the key dim:
//   * NBT <= 8 (v2t, 64 keys): single pass, score fragments kept live.
//   * NBT  > 8 (t2v, 256 keys): two-pass online softmax (recompute S) to
//     cap register pressure at acc(128) + 1 score frag instead of 256+.
// Then per-row top-3 via sorted-triple shuffle merge -> sim[i,j].
// ---------------------------------------------------------------------------
template <int NA, int NB, int NWAVES, int MT>
__global__ void attn_sim_kernel(const __bf16* __restrict__ Q,
                                const __bf16* __restrict__ K,
                                float* __restrict__ sim) {
    constexpr int NBT = NB / 16;               // key tiles
    constexpr bool TWO_PASS = (NBT > 8);
    const int i = blockIdx.x, j = blockIdx.y;
    const int tid = threadIdx.x;
    const int wave = tid >> 5, lane = tid & 31;
    const int l16 = lane & 15, kh = lane >> 4;

    __shared__ __bf16 lk[NB * 72];             // NB x (64+8) bf16 head slice
    __shared__ float wsum[NWAVES];

    f32x8 acc[MT][NBT];
#pragma unroll
    for (int mi = 0; mi < MT; ++mi)
#pragma unroll
        for (int nt = 0; nt < NBT; ++nt)
#pragma unroll
            for (int r = 0; r < 8; ++r) acc[mi][nt][r] = 0.f;

    for (int h = 0; h < HEADS; ++h) {
        // stage K[j, :, h*64 : h*64+64] -> LDS via async-LDS b128 copies
        // (bf16 source, no conversion needed; rows padded to 72 elements).
        // Loop trip counts are uniform across the block (NB*8 % threads == 0)
        // so EXEC is all-ones for every issued async op.
        for (int idx = tid; idx < NB * 8; idx += NWAVES * 32) {
            int row = idx >> 3, ch = idx & 7;
            unsigned lds_off = (unsigned)(uintptr_t)(&lk[row * 72 + ch * 8]);
            const __bf16* src = K + ((size_t)j * NB + row) * DIM
                                + h * HD + ch * 8;
            asm volatile("global_load_async_to_lds_b128 %0, %1, off"
                         :: "v"(lds_off), "v"(src) : "memory");
        }
        asm volatile("s_wait_asynccnt 0x0" ::: "memory");
        __syncthreads();

#pragma unroll
        for (int mi = 0; mi < MT; ++mi) {
            const int rowbase = (wave + mi * NWAVES) * 16;
            const __bf16* qb = Q + ((size_t)i * NA + rowbase) * DIM + h * HD;
            __builtin_prefetch(qb, 0, 0);
            const bf16x16 a0 = load_frag(qb, DIM, l16, 0, kh);
            const bf16x16 a1 = load_frag(qb, DIM, l16, 32, kh);

            // ---- pass 1: per-lane online softmax stats over own columns ---
            f32x8 sf[TWO_PASS ? 1 : NBT];      // raw scores (1-pass mode only)
            float m[8], s[8];
#pragma unroll
            for (int r = 0; r < 8; ++r) { m[r] = -INFINITY; s[r] = 0.f; }
#pragma unroll
            for (int nt = 0; nt < NBT; ++nt) {
                bf16x16 b0 = load_frag(lk, 72, nt * 16 + l16, 0, kh);
                bf16x16 b1 = load_frag(lk, 72, nt * 16 + l16, 32, kh);
                f32x8 sc;
#pragma unroll
                for (int r = 0; r < 8; ++r) sc[r] = 0.f;
                sc = wmma_bf16(a0, b0, sc);
                sc = wmma_bf16(a1, b1, sc);
                if constexpr (!TWO_PASS) sf[nt] = sc;
#pragma unroll
                for (int r = 0; r < 8; ++r) {
                    float v  = sc[r] * SCALE;
                    float mn = fmaxf(m[r], v);
                    s[r] = s[r] * __expf(m[r] - mn) + __expf(v - mn);
                    m[r] = mn;
                }
            }
            // merge (m,s) across the 16-lane half that shares each row
#pragma unroll
            for (int mask = 1; mask <= 8; mask <<= 1) {
#pragma unroll
                for (int r = 0; r < 8; ++r) {
                    float mo = __shfl_xor(m[r], mask, 32);
                    float so = __shfl_xor(s[r], mask, 32);
                    float mn = fmaxf(m[r], mo);
                    s[r] = s[r] * __expf(m[r] - mn) + so * __expf(mo - mn);
                    m[r] = mn;
                }
            }
            float w[8];
#pragma unroll
            for (int r = 0; r < 8; ++r) w[r] = 0.25f / s[r]; // 1/H head mean

            // ---- pass 2: normalized probabilities into head-mean acc ------
            if constexpr (TWO_PASS) {
#pragma unroll
                for (int nt = 0; nt < NBT; ++nt) {
                    bf16x16 b0 = load_frag(lk, 72, nt * 16 + l16, 0, kh);
                    bf16x16 b1 = load_frag(lk, 72, nt * 16 + l16, 32, kh);
                    f32x8 sc;
#pragma unroll
                    for (int r = 0; r < 8; ++r) sc[r] = 0.f;
                    sc = wmma_bf16(a0, b0, sc);
                    sc = wmma_bf16(a1, b1, sc);
#pragma unroll
                    for (int r = 0; r < 8; ++r)
                        acc[mi][nt][r] += __expf(sc[r] * SCALE - m[r]) * w[r];
                }
            } else {
#pragma unroll
                for (int nt = 0; nt < NBT; ++nt)
#pragma unroll
                    for (int r = 0; r < 8; ++r)
                        acc[mi][nt][r] +=
                            __expf(sf[nt][r] * SCALE - m[r]) * w[r];
            }
        }
        __syncthreads();   // protect lk before next head restage
    }

    // ---- per-row top-3 of head-mean attention, then accumulate ------------
    float partial = 0.f;
#pragma unroll
    for (int mi = 0; mi < MT; ++mi) {
#pragma unroll
        for (int r = 0; r < 8; ++r) {
            Top3 t; t3_init(t);
#pragma unroll
            for (int nt = 0; nt < NBT; ++nt) t3_insert(t, acc[mi][nt][r]);
#pragma unroll
            for (int mask = 1; mask <= 8; mask <<= 1) {
                float oa = __shfl_xor(t.a, mask, 32);
                float ob = __shfl_xor(t.b, mask, 32);
                float oc = __shfl_xor(t.c, mask, 32);
                t3_insert(t, oa); t3_insert(t, ob); t3_insert(t, oc);
            }
            partial += t.a + t.b + t.c;   // rows r (half 0) / r+8 (half 1)
        }
    }
    // lanes within a half hold identical values; combine the two halves
    float p0  = __shfl(partial, 0, 32);
    float p16 = __shfl(partial, 16, 32);
    if (lane == 0) wsum[wave] = p0 + p16;
    __syncthreads();
    if (tid == 0) {
        float tot = 0.f;
        for (int w2 = 0; w2 < NWAVES; ++w2) tot += wsum[w2];
        sim[i * BATCH + j] = 100.f * tot / (3.f * (float)NA);
    }
}

// ---------------------------------------------------------------------------
// Kernel 3: contrastive loss from the two 32x32 sim matrices. One wave;
// lane i handles row i of each direction; full-wave shuffle reduction.
// ---------------------------------------------------------------------------
__device__ __forceinline__ float row_loss(const float* sim, int i) {
    float pos = sim[i * BATCH + i];
    Top3 t; t3_init(t);
    for (int j2 = 0; j2 < BATCH; ++j2)
        if (j2 != i) t3_insert(t, sim[i * BATCH + j2]);
    const float invT = 1.0f / TEMP;
    float x0 = pos * invT, x1 = t.a * invT, x2 = t.b * invT, x3 = t.c * invT;
    float mx  = fmaxf(fmaxf(x0, x1), fmaxf(x2, x3));
    float lse = mx + logf(__expf(x0 - mx) + __expf(x1 - mx) +
                          __expf(x2 - mx) + __expf(x3 - mx));
    return lse - x0;     // -log_softmax[0]
}

__global__ void loss_kernel(const float* __restrict__ simV,
                            const float* __restrict__ simT,
                            float* __restrict__ out) {
    const int lane = threadIdx.x;    // 32 threads = 1 wave
    float lv = row_loss(simV, lane);
    float lt = row_loss(simT, lane);
#pragma unroll
    for (int mask = 1; mask <= 16; mask <<= 1) {
        lv += __shfl_xor(lv, mask, 32);
        lt += __shfl_xor(lt, mask, 32);
    }
    if (lane == 0)
        out[0] = 0.5f * (lv / (float)BATCH) + 0.5f * (lt / (float)BATCH);
}

// ---------------------------------------------------------------------------
extern "C" void kernel_launch(void* const* d_in, const int* in_sizes, int n_in,
                              void* d_out, int out_size, void* d_ws,
                              size_t ws_size, hipStream_t stream) {
    const float* lang = (const float*)d_in[0];   // (32, 64, 256)  f32
    const float* vis  = (const float*)d_in[1];   // (32, 256, 256) f32
    const float* W    = (const float*)d_in[2];   // (768, 256)     f32
    const float* bias = (const float*)d_in[3];   // (768,)         f32
    float* out = (float*)d_out;

    // workspace layout (all 256B aligned)
    char* ws = (char*)d_ws;
    size_t off = 0;
    __bf16* Wbf = (__bf16*)(ws + off); off += (size_t)2 * DIM * DIM * 2;     // 256 KB
    __bf16* Qv  = (__bf16*)(ws + off); off += (size_t)BATCH * NVIS  * DIM * 2;
    __bf16* Kv  = (__bf16*)(ws + off); off += (size_t)BATCH * NVIS  * DIM * 2;
    __bf16* Ql  = (__bf16*)(ws + off); off += (size_t)BATCH * NLANG * DIM * 2;
    __bf16* Kl  = (__bf16*)(ws + off); off += (size_t)BATCH * NLANG * DIM * 2;
    float* simV = (float*)(ws + off);  off += (size_t)BATCH * BATCH * 4;
    float* simT = (float*)(ws + off);  off += (size_t)BATCH * BATCH * 4;
    (void)ws_size; (void)in_sizes; (void)n_in; (void)out_size;

    // 0) weights -> bf16
    {
        int n = 2 * DIM * DIM;
        cvt_weights_kernel<<<(n + 255) / 256, 256, 0, stream>>>(W, Wbf, n);
    }
    // 1) projections (512 vis tiles + 128 lang tiles)
    proj_kernel<<<640, 128, 0, stream>>>(lang, vis, bias, Wbf, Qv, Kv, Ql, Kl);
    // 2) pairwise sims: v2t (Q=vis 256 rows, K=lang 64 keys) and t2v
    attn_sim_kernel<NVIS,  NLANG, 8, 2><<<dim3(BATCH, BATCH), 256, 0, stream>>>(Qv, Kl, simV);
    attn_sim_kernel<NLANG, NVIS,  4, 1><<<dim3(BATCH, BATCH), 128, 0, stream>>>(Ql, Kv, simT);
    // 3) scalar loss
    loss_kernel<<<1, 32, 0, stream>>>(simV, simT, out);
}
